// GT_10084583211430
// MI455X (gfx1250) — matrix operand order repaired and checked by reading, not compiled, not run on hardware
//
#include <hip/hip_runtime.h>
#include <hip/hip_bf16.h>

// ---------------------------------------------------------------------------
// Graph Transformer (3x TransformerConv + LN/ReLU + weighted pooling)
// Dense projections: v_wmma_f32_16x16x32_f16, staged via async global->LDS
// (global_load_async_to_lds_b128 + s_wait_asynccnt). Edge softmax/aggregation
// via atomics (segment max/sum over random dst).
// ---------------------------------------------------------------------------

#define NNODES 50000
#define NEDGES 800000
#define NGRAPH 32
#define HEADD  64      // per-head dim for all layers

typedef __attribute__((ext_vector_type(16))) _Float16 v16h;
typedef __attribute__((ext_vector_type(8)))  _Float16 v8h;
typedef __attribute__((ext_vector_type(8)))  float    v8f;

// order-preserving float<->uint for atomicMax-based segment max
__device__ __forceinline__ unsigned ordF(float f) {
    unsigned b = __float_as_uint(f);
    return (b & 0x80000000u) ? ~b : (b | 0x80000000u);
}
__device__ __forceinline__ float unordF(unsigned u) {
    unsigned b = (u & 0x80000000u) ? (u & 0x7FFFFFFFu) : ~u;
    return __uint_as_float(b);
}

// low 32 bits of a generic pointer into __shared__ = LDS byte offset
__device__ __forceinline__ unsigned ldsOff(const void* p) {
    return (unsigned)(unsigned long long)p;
}
// async DMA 16B global -> LDS, per-lane addresses, constant byte offset
#define ASYNC_B128(lds, gaddr, OFF)                                          \
    asm volatile("global_load_async_to_lds_b128 %0, %1, off offset:" #OFF    \
                 :: "v"(lds), "v"(gaddr) : "memory")
#define WAIT_ASYNC() asm volatile("s_wait_asynccnt 0x0" ::: "memory")

// ----------------------------- elementwise utils ---------------------------

__global__ void fill_f32_k(float* p, float v, long n) {
    long i = (long)blockIdx.x * blockDim.x + threadIdx.x;
    if (i < n) p[i] = v;
}
__global__ void fill_u32_k(unsigned* p, unsigned v, long n) {
    long i = (long)blockIdx.x * blockDim.x + threadIdx.x;
    if (i < n) p[i] = v;
}
__global__ void f32_to_f16_k(const float* __restrict__ in, _Float16* __restrict__ out, long n) {
    long i = (long)blockIdx.x * blockDim.x + threadIdx.x;
    if (i < n) out[i] = (_Float16)in[i];
}
// W [K,Nc] fp32 row-major  ->  Wt [Nc,K] f16 row-major (B^T for the GEMM)
__global__ void wtrans_k(const float* __restrict__ W, _Float16* __restrict__ Wt, int K, int Nc) {
    long i = (long)blockIdx.x * blockDim.x + threadIdx.x;
    if (i >= (long)K * Nc) return;
    int k = (int)(i / Nc), n = (int)(i % Nc);
    Wt[(size_t)n * K + k] = (_Float16)W[i];
}

// ------------------------------- WMMA GEMM ---------------------------------
// C[M,Nc] = A[M,K](f16) * B[K,Nc](f16, given as Bt=[Nc,K]) + bias
// block = 256 threads (8 waves), tile 128x64, K-step 64 (2 WMMA k-slices).

#define TM 128
#define TN 64
#define TK 64
#define LDP 72   // padded LDS row stride in halves (144B, 16B-aligned)

union F16Frag { v16h v; v8h p[2]; };

__global__ __launch_bounds__(256)
void gemm_f16_wmma_k(const _Float16* __restrict__ A,
                     const _Float16* __restrict__ Bt,
                     const float* __restrict__ bias,
                     float* __restrict__ C,
                     int M, int K, int Nc)
{
    __shared__ _Float16 As[TM][LDP];
    __shared__ _Float16 Bs[TN][LDP];

    const int tid   = threadIdx.x;
    const int wave  = tid >> 5;
    const int lane  = tid & 31;
    const int laneL = lane & 15;
    const bool hi   = lane >= 16;
    const int mBase = blockIdx.x * TM;
    const int nBase = blockIdx.y * TN;

    // A staging: thread -> (row, 32-half chunk); OOB rows clamped (masked at store)
    const int aRow  = tid >> 1;
    const int aCol  = (tid & 1) * 32;
    const int aGRow = min(mBase + aRow, M - 1);
    const _Float16* aSrc = A + (size_t)aGRow * K + aCol;
    const unsigned  aLds = ldsOff(&As[aRow][aCol]);
    // B staging: thread -> (col, 16-half chunk)
    const int bRow  = tid >> 2;
    const int bCol  = (tid & 3) * 16;
    const _Float16* bSrc = Bt + (size_t)(nBase + bRow) * K + bCol;
    const unsigned  bLds = ldsOff(&Bs[bRow][bCol]);

    v8f acc[4] = {};

    for (int k0 = 0; k0 < K; k0 += TK) {
        // ---- async DMA this K-tile straight into LDS (no VGPR round-trip)
        {
            unsigned long long ga = (unsigned long long)(aSrc + k0);
            ASYNC_B128(aLds, ga, 0);
            ASYNC_B128(aLds, ga, 16);
            ASYNC_B128(aLds, ga, 32);
            ASYNC_B128(aLds, ga, 48);
            unsigned long long gb = (unsigned long long)(bSrc + k0);
            ASYNC_B128(bLds, gb, 0);
            ASYNC_B128(bLds, gb, 16);
        }
        // ---- prefetch next K-tile toward L2 while this one computes
        if (k0 + TK < K) {
            __builtin_prefetch(aSrc + k0 + TK, 0, 3);
            __builtin_prefetch(bSrc + k0 + TK, 0, 3);
        }
        WAIT_ASYNC();      // this wave's DMAs landed in LDS
        __syncthreads();   // all waves' DMAs landed

        // ---- A fragments (ISA 16-bit A 16x32 layout), two k-slices
        const int arow = wave * 16 + laneL;
        const int klo  = hi ? 8 : 0;
        F16Frag af[2];
        af[0].p[0] = *(const v8h*)&As[arow][klo];
        af[0].p[1] = *(const v8h*)&As[arow][16 + klo];
        af[1].p[0] = *(const v8h*)&As[arow][32 + klo];
        af[1].p[1] = *(const v8h*)&As[arow][48 + klo];

        // ---- B fragments, then back-to-back WMMAs
        const int bk = hi ? 8 : 0;
        F16Frag bf[8];
#pragma unroll
        for (int t = 0; t < 4; ++t) {
            const int bcol = t * 16 + laneL;
#pragma unroll
            for (int ks = 0; ks < 2; ++ks) {
                bf[t * 2 + ks].p[0] = *(const v8h*)&Bs[bcol][ks * 32 + bk];
                bf[t * 2 + ks].p[1] = *(const v8h*)&Bs[bcol][ks * 32 + bk + 16];
            }
        }
#pragma unroll
        for (int t = 0; t < 4; ++t) {
            acc[t] = __builtin_amdgcn_wmma_f32_16x16x32_f16(
                false, af[0].v, false, bf[t * 2].v, (short)0, acc[t], false, false);
            acc[t] = __builtin_amdgcn_wmma_f32_16x16x32_f16(
                false, af[1].v, false, bf[t * 2 + 1].v, (short)0, acc[t], false, false);
        }
        __syncthreads();
    }

    // ---- store per C/D layout: vgpr r -> M=r (lanes<16) / M=r+8 (lanes>=16)
#pragma unroll
    for (int t = 0; t < 4; ++t) {
        const int n = nBase + t * 16 + laneL;
        const float bv = bias ? bias[n] : 0.0f;
#pragma unroll
        for (int r = 0; r < 8; ++r) {
            const int m = mBase + wave * 16 + r + (hi ? 8 : 0);
            if (m < M) C[(size_t)m * Nc + n] = acc[t][r] + bv;
        }
    }
}

// ----------------------------- edge pipeline -------------------------------

// alpha_raw = dot(q[dst], k[src] + ea*We) / sqrt(d); segment max via atomicMax
__global__ void alpha_max_k(const float* __restrict__ q, const float* __restrict__ k,
                            const int* __restrict__ ei, const float* __restrict__ ea,
                            const float* __restrict__ We,
                            float* __restrict__ alpha, unsigned* __restrict__ amax,
                            int E, int h, int hw, float scale)
{
    int e = blockIdx.x * blockDim.x + threadIdx.x;
    if (e >= E) return;
    const int src = ei[e], dst = ei[E + e];
    const float a = ea[e];
    for (int hd = 0; hd < h; ++hd) {
        const float4* qp = (const float4*)(q  + (size_t)dst * hw + hd * HEADD);
        const float4* kp = (const float4*)(k  + (size_t)src * hw + hd * HEADD);
        const float4* wp = (const float4*)(We + hd * HEADD);
        float s = 0.f;
#pragma unroll 4
        for (int j = 0; j < HEADD / 4; ++j) {
            float4 qv = qp[j], kv = kp[j], wv = wp[j];
            s += qv.x * (kv.x + a * wv.x) + qv.y * (kv.y + a * wv.y)
               + qv.z * (kv.z + a * wv.z) + qv.w * (kv.w + a * wv.w);
        }
        s *= scale;
        alpha[(size_t)e * h + hd] = s;
        atomicMax(&amax[(size_t)dst * h + hd], ordF(s));
    }
}

// ex = exp(raw - amax[dst]); denom += ex
__global__ void exp_sum_k(const int* __restrict__ ei, float* __restrict__ alpha,
                          const unsigned* __restrict__ amax, float* __restrict__ denom,
                          int E, int h)
{
    int e = blockIdx.x * blockDim.x + threadIdx.x;
    if (e >= E) return;
    const int dst = ei[E + e];
    for (int hd = 0; hd < h; ++hd) {
        const size_t ia = (size_t)e * h + hd;
        float ex = __expf(alpha[ia] - unordF(amax[(size_t)dst * h + hd]));
        alpha[ia] = ex;
        atomicAdd(&denom[(size_t)dst * h + hd], ex);
    }
}

// alpha = ex / (denom[dst] + eps)
__global__ void norm_alpha_k(const int* __restrict__ ei, float* __restrict__ alpha,
                             const float* __restrict__ denom, int E, int h)
{
    long i = (long)blockIdx.x * blockDim.x + threadIdx.x;
    if (i >= (long)E * h) return;
    int e = (int)(i / h), hd = (int)(i % h);
    int dst = ei[E + e];
    alpha[i] = alpha[i] / (denom[(size_t)dst * h + hd] + 1e-16f);
}

// out[dst] += (v[src] + ea*We) * alpha   (out pre-loaded with x@Ws + bs)
__global__ void agg_k(const float* __restrict__ v, const int* __restrict__ ei,
                      const float* __restrict__ ea, const float* __restrict__ We,
                      const float* __restrict__ alpha, float* __restrict__ out,
                      int E, int h, int hw)
{
    long i = (long)blockIdx.x * blockDim.x + threadIdx.x;
    if (i >= (long)E * hw) return;
    int e = (int)(i / hw), j = (int)(i % hw);
    int src = ei[e], dst = ei[E + e];
    float al = alpha[(size_t)e * h + (j / HEADD)];
    float m  = (v[(size_t)src * hw + j] + ea[e] * We[j]) * al;
    atomicAdd(&out[(size_t)dst * hw + j], m);
}

// ------------------------------- epilogues ---------------------------------

__global__ void ln_relu_k(const float* __restrict__ in, const float* __restrict__ g,
                          const float* __restrict__ b, float* __restrict__ out,
                          _Float16* __restrict__ outh, int n, int hw)
{
    int row = blockIdx.x * blockDim.x + threadIdx.x;
    if (row >= n) return;
    const float* p = in + (size_t)row * hw;
    float mu = 0.f;
    for (int j = 0; j < hw; ++j) mu += p[j];
    mu /= hw;
    float var = 0.f;
    for (int j = 0; j < hw; ++j) { float t = p[j] - mu; var += t * t; }
    var /= hw;
    float rs = rsqrtf(var + 1e-5f);
    float* o = out + (size_t)row * hw;
    _Float16* oh = outh + (size_t)row * hw;
    for (int j = 0; j < hw; ++j) {
        float y = (p[j] - mu) * rs * g[j] + b[j];
        y = fmaxf(y, 0.f);
        o[j] = y;
        oh[j] = (_Float16)y;
    }
}

// x_res = x + relu(out3)
__global__ void relu_res_k(const float* __restrict__ x, const float* __restrict__ o3,
                           float* __restrict__ xres, long n)
{
    long i = (long)blockIdx.x * blockDim.x + threadIdx.x;
    if (i < n) xres[i] = x[i] + fmaxf(o3[i], 0.f);
}

// am = mean_h(a3); node_att[src]+=am; node_att[dst]+=am
__global__ void am_k(const float* __restrict__ a3, const int* __restrict__ ei,
                     float* __restrict__ am, float* __restrict__ att, int E)
{
    int e = blockIdx.x * blockDim.x + threadIdx.x;
    if (e >= E) return;
    const float* a = a3 + (size_t)e * 6;
    float m = (a[0] + a[1] + a[2] + a[3] + a[4] + a[5]) * (1.f / 6.f);
    am[e] = m;
    atomicAdd(&att[ei[e]], m);
    atomicAdd(&att[ei[E + e]], m);
}

__global__ void pool_sum_k(const float* __restrict__ xres, const float* __restrict__ att,
                           const int* __restrict__ batch, float* __restrict__ sums, long n)
{
    long i = (long)blockIdx.x * blockDim.x + threadIdx.x;
    if (i >= n) return;
    int node = (int)(i / 384), j = (int)(i % 384);
    atomicAdd(&sums[(size_t)batch[node] * 384 + j], xres[i] * att[node]);
}
__global__ void pool_cnt_k(const int* __restrict__ batch, float* __restrict__ counts, int n)
{
    int i = blockIdx.x * blockDim.x + threadIdx.x;
    if (i < n) atomicAdd(&counts[batch[i]], 1.0f);
}
__global__ void pool_div_k(const float* __restrict__ sums, const float* __restrict__ counts,
                           float* __restrict__ pooled, int g)
{
    int i = blockIdx.x * blockDim.x + threadIdx.x;
    if (i >= g * 384) return;
    pooled[i] = sums[i] / fmaxf(counts[i / 384], 1.0f);
}

// ------------------------------- launcher ----------------------------------

static inline size_t ALN(size_t x) { return (x + 255) & ~size_t(255); }

extern "C" void kernel_launch(void* const* d_in, const int* in_sizes, int n_in,
                              void* d_out, int out_size, void* d_ws, size_t ws_size,
                              hipStream_t stream)
{
    const int N = NNODES, E = NEDGES, G = NGRAPH;
    const float* x   = (const float*)d_in[0];
    const int*   ei  = (const int*)d_in[1];
    const float* ea  = (const float*)d_in[2];
    const int*   bat = (const int*)d_in[3];
    // per-layer params: Wq,Wk,Wv,Ws,We,bq,bk,bv,bs starting at 4, 13, 22
    const float* ln1g = (const float*)d_in[31];
    const float* ln1b = (const float*)d_in[32];
    const float* ln2g = (const float*)d_in[33];
    const float* ln2b = (const float*)d_in[34];

    const int din[3] = {384, 256, 256};
    const int hw [3] = {256, 256, 384};
    const int hds[3] = {4, 4, 6};

    // ---- workspace carve-up
    char* w = (char*)d_ws;  size_t off = 0;
    auto take = [&](size_t bytes) { void* p = w + off; off = ALN(off + bytes); return p; };
    float*    qb   = (float*)take((size_t)N * 384 * 4);
    float*    kb   = (float*)take((size_t)N * 384 * 4);
    float*    vb   = (float*)take((size_t)N * 384 * 4);
    float*    outb = (float*)take((size_t)N * 384 * 4);
    float*    x1   = (float*)take((size_t)N * 256 * 4);
    float*    x2   = (float*)take((size_t)N * 256 * 4);
    _Float16* xh   = (_Float16*)take((size_t)N * 384 * 2);
    _Float16* x1h  = (_Float16*)take((size_t)N * 256 * 2);
    _Float16* x2h  = (_Float16*)take((size_t)N * 256 * 2);
    _Float16* wt   = (_Float16*)take((size_t)1048576 * 2);   // all 12 transposed f16 weights
    unsigned* amax = (unsigned*)take((size_t)N * 6 * 4);
    float*    den  = (float*)take((size_t)N * 6 * 4);
    float*    sums = (float*)take((size_t)G * 384 * 4);
    float*    cnts = (float*)take((size_t)G * 4);
    (void)ws_size; (void)in_sizes; (void)n_in; (void)out_size;

    // ---- output carve-up (flat, in return order)
    float* out = (float*)d_out;
    long o = 0;
    float* pooled = out + o;            o += (long)G * 384;
    float* a_out[3];
    a_out[0] = out + o;                 o += (long)E * 4;
    a_out[1] = out + o;                 o += (long)E * 4;
    a_out[2] = out + o;                 o += (long)E * 6;
    float* am_out  = out + o;           o += E;
    float* att_out = out + o;           // [N]

    const int T = 256;
    auto cdiv = [](long a, long b) { return (int)((a + b - 1) / b); };

    // ---- transpose+convert all weights to f16 [Nc][K]
    _Float16* wt_ptr[3][4];
    {
        size_t woff = 0;
        for (int L = 0; L < 3; ++L)
            for (int m = 0; m < 4; ++m) {
                wt_ptr[L][m] = wt + woff;
                const float* W = (const float*)d_in[4 + 9 * L + m];
                long n = (long)din[L] * hw[L];
                wtrans_k<<<cdiv(n, T), T, 0, stream>>>(W, wt_ptr[L][m], din[L], hw[L]);
                woff += (size_t)n;
            }
    }
    // ---- x -> f16
    f32_to_f16_k<<<cdiv((long)N * 384, T), T, 0, stream>>>(x, xh, (long)N * 384);

    const _Float16* act_h[3] = {xh, x1h, x2h};

    for (int L = 0; L < 3; ++L) {
        const int K = din[L], HW = hw[L], H = hds[L];
        const float* We = (const float*)d_in[4 + 9 * L + 4];
        const float* bq = (const float*)d_in[4 + 9 * L + 5];
        const float* bk = (const float*)d_in[4 + 9 * L + 6];
        const float* bv = (const float*)d_in[4 + 9 * L + 7];
        const float* bs = (const float*)d_in[4 + 9 * L + 8];

        dim3 gg(cdiv(N, TM), HW / TN);
        gemm_f16_wmma_k<<<gg, 256, 0, stream>>>(act_h[L], wt_ptr[L][0], bq, qb,   N, K, HW);
        gemm_f16_wmma_k<<<gg, 256, 0, stream>>>(act_h[L], wt_ptr[L][1], bk, kb,   N, K, HW);
        gemm_f16_wmma_k<<<gg, 256, 0, stream>>>(act_h[L], wt_ptr[L][2], bv, vb,   N, K, HW);
        gemm_f16_wmma_k<<<gg, 256, 0, stream>>>(act_h[L], wt_ptr[L][3], bs, outb, N, K, HW); // skip path

        fill_u32_k<<<cdiv((long)N * H, T), T, 0, stream>>>(amax, 0u, (long)N * H);
        fill_f32_k<<<cdiv((long)N * H, T), T, 0, stream>>>(den, 0.f, (long)N * H);

        alpha_max_k<<<cdiv(E, T), T, 0, stream>>>(qb, kb, ei, ea, We, a_out[L], amax,
                                                  E, H, HW, 0.125f);
        exp_sum_k<<<cdiv(E, T), T, 0, stream>>>(ei, a_out[L], amax, den, E, H);
        norm_alpha_k<<<cdiv((long)E * H, T), T, 0, stream>>>(ei, a_out[L], den, E, H);
        agg_k<<<cdiv((long)E * HW, T), T, 0, stream>>>(vb, ei, ea, We, a_out[L], outb,
                                                       E, H, HW);

        if (L == 0)
            ln_relu_k<<<cdiv(N, T), T, 0, stream>>>(outb, ln1g, ln1b, x1, x1h, N, HW);
        else if (L == 1)
            ln_relu_k<<<cdiv(N, T), T, 0, stream>>>(outb, ln2g, ln2b, x2, x2h, N, HW);
        else
            relu_res_k<<<cdiv((long)N * 384, T), T, 0, stream>>>(x, outb, qb /*reuse as x_res*/,
                                                                 (long)N * 384);
    }

    // ---- pooling: node attention, weighted graph means
    fill_f32_k<<<cdiv((long)N, T), T, 0, stream>>>(att_out, 0.f, N);
    fill_f32_k<<<cdiv((long)G * 384, T), T, 0, stream>>>(sums, 0.f, (long)G * 384);
    fill_f32_k<<<1, T, 0, stream>>>(cnts, 0.f, G);

    am_k<<<cdiv(E, T), T, 0, stream>>>(a_out[2], ei, am_out, att_out, E);
    pool_sum_k<<<cdiv((long)N * 384, T), T, 0, stream>>>(qb, att_out, bat, sums, (long)N * 384);
    pool_cnt_k<<<cdiv(N, T), T, 0, stream>>>(bat, cnts, N);
    pool_div_k<<<cdiv((long)G * 384, T), T, 0, stream>>>(sums, cnts, pooled, G);
}
